// MultiHeadAttention_267
// MI455X (gfx1250) — matrix operand order repaired and checked by reading, not compiled, not run on hardware
//
#include <hip/hip_runtime.h>
#include <hip/hip_bf16.h>

typedef __attribute__((ext_vector_type(16))) __bf16 v16bf;
typedef __attribute__((ext_vector_type(8)))  __bf16 v8bf;
typedef __attribute__((ext_vector_type(8)))  float  v8f;

#define NEGVAL (-1e9f)

// ---- fragment loaders (CDNA5 WMMA 16-bit layouts, ISA 7.12.2) ----
// A 16x32 bf16: lane(0-15)=row M, kbase = (lane>=16)?8:0 ; VGPR pairs hold
// K = kbase+{0..7} and kbase+16+{0..7}.
__device__ inline v16bf load_a_frag(const __bf16* p) {
    v8bf lo = *(const v8bf*)(p);        // K = kbase + 0..7
    v8bf hi = *(const v8bf*)(p + 16);   // K = kbase + 16..23
    v16bf r;
#pragma unroll
    for (int i = 0; i < 8; ++i) { r[i] = lo[i]; r[i + 8] = hi[i]; }
    return r;
}
// B 32x16 bf16 (stored column-major = rows of K): lane(0-15)=col N,
// kbase = (lane>=16)?8:0 ; holds K = kbase+{0..7} and kbase+16+{0..7} (matches A).
__device__ inline v16bf load_b_frag(const __bf16* p) {
    v8bf lo = *(const v8bf*)(p);
    v8bf hi = *(const v8bf*)(p + 16);
    v16bf r;
#pragma unroll
    for (int i = 0; i < 8; ++i) { r[i] = lo[i]; r[i + 8] = hi[i]; }
    return r;
}

__device__ inline v8f wmma_bf16(v16bf a, v16bf b, v8f c) {
    return __builtin_amdgcn_wmma_f32_16x16x32_bf16(
        false, a, false, b, (short)0, c, false, false);
}

// ---------------- phase 0: fp32 -> bf16 ----------------
__global__ void cvt_f32_bf16(const float* __restrict__ in, __bf16* __restrict__ out, int n) {
    int i = blockIdx.x * blockDim.x + threadIdx.x;
    if (i < n) out[i] = (__bf16)in[i];
}

// ---------------- generic bf16 WMMA GEMM: C[M,N] = A[M,K] @ Bcm[N,K]^T + bias ----------------
// A row-major (lda=K), Bcm holds B column-major (i.e. weight row-major: Bcm[n*K+k]).
// 4 waves/block, each wave computes a 16(M) x 64(N) strip.
__global__ void gemm_bf16_wmma(const __bf16* __restrict__ A,
                               const __bf16* __restrict__ Bcm,
                               const float* __restrict__ bias,
                               void* __restrict__ Cout,
                               int M, int N, int K, int c_is_f32) {
    const int lane = threadIdx.x & 31;
    const int wv   = threadIdx.x >> 5;
    const int mtile = (blockIdx.y * 4 + wv) * 16;
    const int ntile = blockIdx.x * 64;
    const int row16 = lane & 15;
    const int kh    = (lane >> 4) & 1;
    const int kbase8  = kh * 8;    // A fragment K base
    const int kbase16 = kh * 16;   // unused; B fragment also uses kbase8

    v8f c[4];
#pragma unroll
    for (int j = 0; j < 4; ++j) {
        float bv = bias ? bias[ntile + 16 * j + row16] : 0.0f;
#pragma unroll
        for (int r = 0; r < 8; ++r) c[j][r] = bv;
    }

    const __bf16* arow = A + (size_t)(mtile + row16) * K;
    for (int k0 = 0; k0 < K; k0 += 32) {
        v16bf a = load_a_frag(arow + k0 + kbase8);
#pragma unroll
        for (int j = 0; j < 4; ++j) {
            const __bf16* bp = Bcm + (size_t)(ntile + 16 * j + row16) * K + k0 + kbase8;
            v16bf b = load_b_frag(bp);
            c[j] = wmma_bf16(a, b, c[j]);
        }
    }

#pragma unroll
    for (int j = 0; j < 4; ++j) {
        int ncol = ntile + 16 * j + row16;
#pragma unroll
        for (int r = 0; r < 8; ++r) {
            int mrow = mtile + kh * 8 + r;
            if (c_is_f32) ((float*)Cout)[(size_t)mrow * N + ncol] = c[j][r];
            else          ((__bf16*)Cout)[(size_t)mrow * N + ncol] = (__bf16)c[j][r];
        }
    }
}

// ---------------- transpose V per head: Vp[b,s,h*64+d] -> Vt[b,h,d,s] ----------------
__global__ void transpose_v(const __bf16* __restrict__ Vp, __bf16* __restrict__ Vt,
                            int B, int S, int D, int H) {
    size_t i = (size_t)blockIdx.x * blockDim.x + threadIdx.x;
    size_t total = (size_t)B * S * D;
    if (i >= total) return;
    int b = (int)(i / ((size_t)S * D));
    int rem = (int)(i % ((size_t)S * D));
    int s = rem / D;
    int d = rem % D;
    int h = d >> 6, dd = d & 63;
    Vt[(((size_t)b * H + h) * 64 + dd) * S + s] = Vp[i];
}

// ---------------- scores: attn_raw = scale * Q Kt, causal mask fused ----------------
__global__ void scores_wmma(const __bf16* __restrict__ Qp, const __bf16* __restrict__ Kp,
                            float* __restrict__ attn, int S, int D, int H) {
    const int bh = blockIdx.z;
    const int b = bh / H, h = bh % H;
    const int lane = threadIdx.x & 31;
    const int wv   = threadIdx.x >> 5;
    const int mtile = (blockIdx.y * 4 + wv) * 16;
    const int ntile = blockIdx.x * 64;
    const int row16 = lane & 15;
    const int kh = (lane >> 4) & 1;
    const int kbase8 = kh * 8, kbase16 = kh * 16;

    const __bf16* Qb = Qp + (size_t)b * S * D + (size_t)h * 64;
    const __bf16* Kb = Kp + (size_t)b * S * D + (size_t)h * 64;
    float* out = attn + (size_t)bh * S * S;

    v8f c[4];
#pragma unroll
    for (int j = 0; j < 4; ++j)
#pragma unroll
        for (int r = 0; r < 8; ++r) c[j][r] = 0.0f;

    // wave-uniform causal skip: tile entirely above diagonal -> no compute
    if (ntile <= mtile + 15) {
        const __bf16* arow = Qb + (size_t)(mtile + row16) * D;
#pragma unroll
        for (int k0 = 0; k0 < 64; k0 += 32) {
            v16bf a = load_a_frag(arow + k0 + kbase8);
#pragma unroll
            for (int j = 0; j < 4; ++j) {
                const __bf16* bp = Kb + (size_t)(ntile + 16 * j + row16) * D + k0 + kbase8;
                v16bf bfr = load_b_frag(bp);
                c[j] = wmma_bf16(a, bfr, c[j]);
            }
        }
    }

    const float scale = 0.125f;  // 1/sqrt(64)
#pragma unroll
    for (int j = 0; j < 4; ++j) {
        int ncol = ntile + 16 * j + row16;
#pragma unroll
        for (int r = 0; r < 8; ++r) {
            int mrow = mtile + kh * 8 + r;
            float v = (ncol <= mrow) ? c[j][r] * scale : NEGVAL;
            out[(size_t)mrow * S + ncol] = v;
        }
    }
}

// ---------------- row softmax in place ----------------
__global__ void softmax_rows(float* __restrict__ attn, int S) {
    __shared__ float red[256];
    float* row = attn + (size_t)blockIdx.x * S;
    int tid = threadIdx.x;

    float m = -3.4e38f;
    for (int j = tid; j < S; j += 256) m = fmaxf(m, row[j]);
    red[tid] = m; __syncthreads();
    for (int off = 128; off > 0; off >>= 1) {
        if (tid < off) red[tid] = fmaxf(red[tid], red[tid + off]);
        __syncthreads();
    }
    float rmax = red[0];
    __syncthreads();

    float s = 0.0f;
    for (int j = tid; j < S; j += 256) {
        float e = __expf(row[j] - rmax);
        row[j] = e;
        s += e;
    }
    red[tid] = s; __syncthreads();
    for (int off = 128; off > 0; off >>= 1) {
        if (tid < off) red[tid] += red[tid + off];
        __syncthreads();
    }
    float inv = 1.0f / red[0];
    for (int j = tid; j < S; j += 256) row[j] *= inv;
}

// ---------------- ctx = attn @ V  (A fp32->bf16 on the fly, B from Vt contiguous) ----------------
__global__ void ctx_wmma(const float* __restrict__ attn, const __bf16* __restrict__ Vt,
                         __bf16* __restrict__ ctx, int S, int D, int H) {
    const int bh = blockIdx.z;
    const int b = bh / H, h = bh % H;
    const int lane = threadIdx.x & 31;
    const int wv   = threadIdx.x >> 5;
    const int mtile = (blockIdx.y * 4 + wv) * 16;
    const int row16 = lane & 15;
    const int kh = (lane >> 4) & 1;
    const int kbase8 = kh * 8, kbase16 = kh * 16;

    const float* Ab = attn + (size_t)bh * S * S;
    const __bf16* Vb = Vt + (size_t)bh * 64 * S;   // [64][S] contiguous along S

    v8f c[4];
#pragma unroll
    for (int j = 0; j < 4; ++j)
#pragma unroll
        for (int r = 0; r < 8; ++r) c[j][r] = 0.0f;

    const float* arow = Ab + (size_t)(mtile + row16) * S;
    const int kmax = mtile + 16;   // causal: attn is exactly 0 beyond the diagonal
    for (int k0 = 0; k0 < kmax; k0 += 32) {
        const float* ap = arow + k0 + kbase8;
        v16bf a;
#pragma unroll
        for (int i = 0; i < 8; ++i) {
            a[i]     = (__bf16)ap[i];
            a[i + 8] = (__bf16)ap[i + 16];
        }
#pragma unroll
        for (int j = 0; j < 4; ++j) {
            const __bf16* bp = Vb + (size_t)(16 * j + row16) * S + k0 + kbase8;
            v16bf bfr = load_b_frag(bp);
            c[j] = wmma_bf16(a, bfr, c[j]);
        }
    }

#pragma unroll
    for (int j = 0; j < 4; ++j) {
        int ncol = 16 * j + row16;
#pragma unroll
        for (int r = 0; r < 8; ++r) {
            int mrow = mtile + kh * 8 + r;
            ctx[((size_t)b * S + mrow) * D + (size_t)h * 64 + ncol] = (__bf16)c[j][r];
        }
    }
}

extern "C" void kernel_launch(void* const* d_in, const int* in_sizes, int n_in,
                              void* d_out, int out_size, void* d_ws, size_t ws_size,
                              hipStream_t stream) {
    const int B = 2, S = 2048, D = 1024, H = 16;
    const int M = B * S;                       // 4096
    const size_t MD = (size_t)M * D;           // 4 Mi elems
    const size_t DD = (size_t)D * D;           // 1 Mi elems

    const float* q   = (const float*)d_in[0];
    const float* k   = (const float*)d_in[1];
    const float* v   = (const float*)d_in[2];
    // d_in[3] = mask (tril, applied arithmetically in scores_wmma)
    const float* w_q = (const float*)d_in[4];
    const float* b_q = (const float*)d_in[5];
    const float* w_k = (const float*)d_in[6];
    const float* b_k = (const float*)d_in[7];
    const float* w_v = (const float*)d_in[8];
    const float* b_v = (const float*)d_in[9];
    const float* w_o = (const float*)d_in[10];
    const float* b_o = (const float*)d_in[11];

    // workspace layout (bf16)
    __bf16* xq = (__bf16*)d_ws;
    __bf16* xk = xq + MD;
    __bf16* xv = xk + MD;
    __bf16* Qp = xv + MD;
    __bf16* Kp = Qp + MD;
    __bf16* Vp = Kp + MD;
    __bf16* wq = Vp + MD;
    __bf16* wk = wq + DD;
    __bf16* wv = wk + DD;
    __bf16* wo = wv + DD;
    __bf16* Vt  = xk;   // reuse: xk dead after Kp GEMM
    __bf16* ctx = xq;   // reuse: xq dead after Qp GEMM

    float* out  = (float*)d_out;
    float* attn = out + MD;   // tuple output #2: [B,H,S,S]

    const int CT = 256;
    // phase 0: conversions
    cvt_f32_bf16<<<(int)((MD + CT - 1) / CT), CT, 0, stream>>>(q, xq, (int)MD);
    cvt_f32_bf16<<<(int)((MD + CT - 1) / CT), CT, 0, stream>>>(k, xk, (int)MD);
    cvt_f32_bf16<<<(int)((MD + CT - 1) / CT), CT, 0, stream>>>(v, xv, (int)MD);
    cvt_f32_bf16<<<(int)((DD + CT - 1) / CT), CT, 0, stream>>>(w_q, wq, (int)DD);
    cvt_f32_bf16<<<(int)((DD + CT - 1) / CT), CT, 0, stream>>>(w_k, wk, (int)DD);
    cvt_f32_bf16<<<(int)((DD + CT - 1) / CT), CT, 0, stream>>>(w_v, wv, (int)DD);
    cvt_f32_bf16<<<(int)((DD + CT - 1) / CT), CT, 0, stream>>>(w_o, wo, (int)DD);

    // phase 1: Q/K/V projections (y = x @ W^T + b), bf16 out
    dim3 gB(128);
    dim3 gG(D / 64, M / 64);
    gemm_bf16_wmma<<<gG, gB, 0, stream>>>(xq, wq, b_q, (void*)Qp, M, D, D, 0);
    gemm_bf16_wmma<<<gG, gB, 0, stream>>>(xk, wk, b_k, (void*)Kp, M, D, D, 0);
    gemm_bf16_wmma<<<gG, gB, 0, stream>>>(xv, wv, b_v, (void*)Vp, M, D, D, 0);

    // phase 2: transpose V per head
    transpose_v<<<(int)((MD + CT - 1) / CT), CT, 0, stream>>>(Vp, Vt, B, S, D, H);

    // phase 3: scores (scale + causal mask fused) -> attn region of d_out
    scores_wmma<<<dim3(S / 64, S / 64, B * H), 128, 0, stream>>>(Qp, Kp, attn, S, D, H);

    // phase 4: softmax rows in place
    softmax_rows<<<B * H * S, 256, 0, stream>>>(attn, S);

    // phase 5: ctx = attn @ V
    ctx_wmma<<<dim3(1, S / 64, B * H), 128, 0, stream>>>(attn, Vt, ctx, S, D, H);

    // phase 6: out = ctx @ W_o^T + b_o, fp32 out
    gemm_bf16_wmma<<<gG, gB, 0, stream>>>(ctx, wo, b_o, (void*)out, M, D, D, 1);
}
